// VanillaRNN_76957224009810
// MI455X (gfx1250) — matrix operand order repaired and checked by reading, not compiled, not run on hardware
//
#include <hip/hip_runtime.h>
#include <hip/hip_bf16.h>

// ---------------------------------------------------------------------------
// Vanilla RNN forward on gfx1250 (MI455X), bf16 WMMA with f32 accumulate.
//   h: [H=128, Btile] kept in LDS (bf16, transposed, double-buffered)
//   Whh A-fragments: resident in VGPRs across all 512 steps
//   16 x v_wmma_f32_16x16x32_bf16 per wave per timestep
//   sched_group_barrier pipeline: ds_load groups run one K-step ahead of the
//   WMMA groups -> partial s_wait_dscnt instead of full-latency stalls.
// ---------------------------------------------------------------------------

#define SEQ_LEN   512
#define HID       128
#define NCLS      10
#define BATCH     4096
#define BT        64          // batch columns per workgroup
#define TCH       64          // timesteps staged per x-chunk
#define THREADS   256         // 8 wave32
#define HT_STRIDE 136         // padded K-stride of hT (bf16 elems); 272B rows, 16B-aligned

// sched_group_barrier masks
#define SG_MFMA   0x008       // MFMA/WMMA class
#define SG_DSREAD 0x100       // DS read

typedef __attribute__((ext_vector_type(16))) __bf16 v16bf;
typedef __attribute__((ext_vector_type(8)))  __bf16 v8bf;
typedef __attribute__((ext_vector_type(8)))  float  v8f;

union BFrag {
    v16bf v;
    v8bf  h[2];
    __bf16 e[16];
};

__global__ __launch_bounds__(THREADS)
void rnn_wmma_kernel(const float* __restrict__ x,     // [BATCH, SEQ_LEN]
                     const float* __restrict__ Whx,   // [HID, 1]
                     const float* __restrict__ Whh,   // [HID, HID]
                     const float* __restrict__ Wph,   // [NCLS, HID]
                     const float* __restrict__ bh,    // [HID]
                     const float* __restrict__ bp_in, // [NCLS]
                     float* __restrict__ out)         // [BATCH, NCLS]
{
    const int tid    = threadIdx.x;
    const int lane   = tid & 31;
    const int wid    = tid >> 5;          // 0..7 : M-strip of 16 rows
    const int batch0 = blockIdx.x * BT;

    // hT[buf][n][k]: h transposed, bf16, padded stride. Double buffered.
    __shared__ __align__(32) __bf16 hT[2][BT][HT_STRIDE];   // 34816 B
    __shared__ float xs[TCH][BT + 1];                       // 16640 B (padded)

    // ---- init h = 0 in buffer 0 ----
    for (int i = tid; i < BT * HT_STRIDE; i += THREADS)
        (&hT[0][0][0])[i] = (__bf16)0.0f;

    // ---- build resident A-fragments of Whh (bf16) -------------------------
    // 16-bit A 16x32 layout: lanes 0-15 row M=lane, K={0..7,16..23};
    //                        lanes 16-31 row M=lane-16, K={8..15,24..31}.
    const int mrow = (wid << 4) + (lane & 15);
    const int koff = (lane >> 4) << 3;               // 0 or 8
    BFrag afr[4];
#pragma unroll
    for (int kt = 0; kt < 4; ++kt) {
        const float* src = Whh + mrow * HID + kt * 32 + koff;
#pragma unroll
        for (int j = 0; j < 8; ++j) afr[kt].e[j]     = (__bf16)src[j];
#pragma unroll
        for (int j = 0; j < 8; ++j) afr[kt].e[8 + j] = (__bf16)src[16 + j];
    }

    // ---- per-lane row constants for the C/D layout ------------------------
    // acc VGPR r holds (M = mcd + r, N = nb + (lane&15))
    const int mcd = (wid << 4) + ((lane >> 4) << 3);
    float whx_r[8], bh_r[8];
#pragma unroll
    for (int r = 0; r < 8; ++r) {
        whx_r[r] = Whx[mcd + r];
        bh_r[r]  = bh[mcd + r];
    }
    __syncthreads();

    const float TWO_LOG2E = 2.8853900817779268f;     // 2*log2(e)
    const int   nlo   = lane & 15;
    const int   khalf = (lane >> 4) << 3;            // 0 or 8

    // ======================= time loop =====================================
    for (int tc = 0; tc < SEQ_LEN; tc += TCH) {
        // stage x[batch0..+BT)[tc..tc+TCH) into LDS, transposed to xs[t][b]
        __syncthreads();                              // xs reuse fence
        {
            const int row = tid >> 2;                 // 0..63 (batch row)
            const int i0  = (tid & 3) << 4;           // 0,16,32,48 (t offset)
            const float* src = x + (size_t)(batch0 + row) * SEQ_LEN + tc + i0;
#pragma unroll
            for (int j = 0; j < 16; ++j)
                xs[i0 + j][row] = src[j];
        }
        __syncthreads();

#pragma unroll 2                       // make (t&1) compile-time: static LDS bases
        for (int tl = 0; tl < TCH; ++tl) {
            const int t = tc + tl;
            const __bf16* bufR = &hT[t & 1][0][0];
            __bf16*       bufW = &hT[(t + 1) & 1][0][0];

            // ---- preload ALL 16 B fragments (4 N-tiles x 4 K-steps) -------
            BFrag bf[4][4];
#pragma unroll
            for (int nt = 0; nt < 4; ++nt) {
                const __bf16* bcol = bufR + ((nt << 4) + nlo) * HT_STRIDE + khalf;
#pragma unroll
                for (int kt = 0; kt < 4; ++kt) {
                    bf[nt][kt].h[0] = *(const v8bf*)(bcol + kt * 32);
                    bf[nt][kt].h[1] = *(const v8bf*)(bcol + kt * 32 + 16);
                }
            }

            v8f acc[4];
#pragma unroll
            for (int nt = 0; nt < 4; ++nt)
                acc[nt] = (v8f){0.f, 0.f, 0.f, 0.f, 0.f, 0.f, 0.f, 0.f};

            // kt-major: each A fragment feeds 4 consecutive WMMAs
#pragma unroll
            for (int kt = 0; kt < 4; ++kt) {
#pragma unroll
                for (int nt = 0; nt < 4; ++nt) {
                    acc[nt] = __builtin_amdgcn_wmma_f32_16x16x32_bf16(
                            false, afr[kt].v, false, bf[nt][kt].v,
                            (short)0, acc[nt], false, false);
                }
            }

            // ---- epilogue: h_new = tanh(acc + Whx[m]*x_t[n] + bh[m]) ------
#pragma unroll
            for (int nt = 0; nt < 4; ++nt) {
                const int nb = nt << 4;
                const float xv = xs[tl][nb + nlo];
                __bf16* wp = bufW + (nb + nlo) * HT_STRIDE + mcd;
#pragma unroll
                for (int r = 0; r < 8; ++r) {
                    float a = acc[nt][r] + whx_r[r] * xv + bh_r[r];
                    // tanh(a) = 1 - 2/(exp(2a)+1), via v_exp_f32 + v_rcp_f32
                    float e  = __builtin_amdgcn_exp2f(a * TWO_LOG2E);
                    float hn = 1.0f - 2.0f * __builtin_amdgcn_rcpf(e + 1.0f);
                    wp[r] = (__bf16)hn;
                }
            }

            // ---- scheduling pipeline: loads run one K-group ahead ---------
            //  G1: 16 ds reads (kt0+kt1)   G2: 4 wmma (kt0)
            //  G3:  8 ds reads (kt2)       G4: 4 wmma (kt1)
            //  G5:  8 ds reads (kt3)       G6: 8 wmma (kt2+kt3)
            __builtin_amdgcn_sched_group_barrier(SG_DSREAD, 16, 0);
            __builtin_amdgcn_sched_group_barrier(SG_MFMA,    4, 0);
            __builtin_amdgcn_sched_group_barrier(SG_DSREAD,  8, 0);
            __builtin_amdgcn_sched_group_barrier(SG_MFMA,    4, 0);
            __builtin_amdgcn_sched_group_barrier(SG_DSREAD,  8, 0);
            __builtin_amdgcn_sched_group_barrier(SG_MFMA,    8, 0);

            __syncthreads();   // hT[(t+1)&1] complete before next-step reads
        }
    }

    // ======================= projection ====================================
    // final h is in hT[SEQ_LEN & 1] == hT[0];  p[b][c] = Wph[c,:] . h[:,b] + bp
    const __bf16* hf = &hT[SEQ_LEN & 1][0][0];
    for (int idx = tid; idx < BT * NCLS; idx += THREADS) {
        const int b = idx & (BT - 1);
        const int c = idx / BT;
        float s = bp_in[c];
        const float* wrow = Wph + c * HID;
#pragma unroll 4
        for (int k = 0; k < HID; ++k)
            s += wrow[k] * (float)hf[b * HT_STRIDE + k];
        out[(size_t)(batch0 + b) * NCLS + c] = s;
    }
}

extern "C" void kernel_launch(void* const* d_in, const int* in_sizes, int n_in,
                              void* d_out, int out_size, void* d_ws, size_t ws_size,
                              hipStream_t stream)
{
    (void)in_sizes; (void)n_in; (void)d_ws; (void)ws_size; (void)out_size;
    const float* x   = (const float*)d_in[0];
    const float* Whx = (const float*)d_in[1];
    const float* Whh = (const float*)d_in[2];
    const float* Wph = (const float*)d_in[3];
    const float* bh  = (const float*)d_in[4];
    const float* bp  = (const float*)d_in[5];
    float* out = (float*)d_out;

    dim3 grid(BATCH / BT);    // 64 workgroups
    dim3 block(THREADS);      // 8 wave32
    rnn_wmma_kernel<<<grid, block, 0, stream>>>(x, Whx, Whh, Wph, bh, bp, out);
}
